// AttentionSuper_45466523795470
// MI455X (gfx1250) — matrix operand order repaired and checked by reading, compiled, not run on hardware
//
#include <hip/hip_runtime.h>
#include <hip/hip_bf16.h>

// ---------------------------------------------------------------------------
// AttentionSuper on MI455X (gfx1250).  All matmuls -> v_wmma_f32_16x16x32_bf16
// (fp32 accumulate).  All GEMM operands are pre-materialized in global memory
// as bf16, K-contiguous, in the orientation each tile wants, so the GEMM hot
// loop is only:  2x global b128 load + 2x ds_store_b128 (stage)  and
// 2x ds_load_b128 per fragment -> v_wmma.  No conversions, no transposes in
// the hot path.  Double-buffered LDS, one barrier per K-step.
// A-tiles are stored with the ISA 7.12.2 K-permutation [0..7,16..23,8..15,
// 24..31] folded into the LDS store offset so each lane's A-frag is one
// contiguous 32B run; B-tiles are stored n-major/k-contiguous likewise.
// ---------------------------------------------------------------------------

typedef __bf16 bf16_t;
typedef __attribute__((ext_vector_type(16))) __bf16 v16bf;
typedef __attribute__((ext_vector_type(8)))  __bf16 v8bf;
typedef __attribute__((ext_vector_type(4)))  __bf16 v4bf;
typedef __attribute__((ext_vector_type(8)))  float  v8f;

#define NN   197
#define HH   10
#define DD   64
#define CC   640
#define QLD  1920           // qkv row stride (bf16 elems)
#define ALD  224            // padded attn/P row stride, 7*32
#define MT   12608          // B*N = 64*197 token rows
#define SCALE 0.125f        // 64^-0.5

#define APAD 40             // LDS A tile row stride (bf16): 80B, 16B aligned
#define BTP  40             // LDS B tile row stride (bf16)

// ---- workspace layout (bytes, 256-aligned) --------------------------------
static constexpr size_t al256(size_t x) { return (x + 255) & ~(size_t)255; }
static constexpr size_t OFF_QKVBF = 0;                                          // bf16 [12608][1920]
static constexpr size_t OFF_ATTN  = al256(OFF_QKVBF + (size_t)MT * QLD * 2);    // f32  [640][197][224]
static constexpr size_t OFF_PBF   = al256(OFF_ATTN  + (size_t)640 * NN * ALD * 4); // bf16 [640][197][224]
static constexpr size_t OFF_RPK   = al256(OFF_PBF   + (size_t)640 * NN * ALD * 2); // bf16 [197][197][64]
static constexpr size_t OFF_RPVT  = al256(OFF_RPK   + (size_t)NN * NN * DD * 2);   // bf16 [197][64][224]
static constexpr size_t OFF_CTX   = al256(OFF_RPVT  + (size_t)NN * DD * ALD * 2);  // f32  [12608][640]
static constexpr size_t OFF_CTXBF = al256(OFF_CTX   + (size_t)MT * CC * 4);        // bf16 [12608][640]
static constexpr size_t OFF_XBF   = al256(OFF_CTXBF + (size_t)MT * CC * 2);        // bf16 [12608][640]
static constexpr size_t OFF_WQT   = al256(OFF_XBF   + (size_t)MT * CC * 2);        // bf16 [1920][640]
static constexpr size_t OFF_WPT   = al256(OFF_WQT   + (size_t)1920 * CC * 2);      // bf16 [640][640]
static constexpr size_t OFF_VT    = al256(OFF_WPT   + (size_t)CC * CC * 2);        // bf16 [640][64][224]

// ---- fragment load: 16 contiguous bf16 = 2x ds_load_b128 ------------------
__device__ __forceinline__ v16bf ld_frag16(const bf16_t* p) {
  v8bf lo = *(const v8bf*)p;
  v8bf hi = *(const v8bf*)(p + 8);
  return __builtin_shufflevector(lo, hi, 0, 1, 2, 3, 4, 5, 6, 7,
                                 8, 9, 10, 11, 12, 13, 14, 15);
}

__device__ __forceinline__ void mma_tile(const bf16_t* As_, const bf16_t* Bt_,
                                         int wm, int wn, int lane, v8f (&acc)[2][2]) {
  int ah = (lane >> 4) << 4;          // K-half select for both A and B frags
  int ml = lane & 15;
  v16bf a0 = ld_frag16(As_ + (wm * 32 + ml) * APAD + ah);
  v16bf a1 = ld_frag16(As_ + (wm * 32 + 16 + ml) * APAD + ah);
  v16bf b0 = ld_frag16(Bt_ + (wn * 32 + ml) * BTP + ah);
  v16bf b1 = ld_frag16(Bt_ + (wn * 32 + 16 + ml) * BTP + ah);
  acc[0][0] = __builtin_amdgcn_wmma_f32_16x16x32_bf16(false, a0, false, b0, (short)0, acc[0][0], false, false);
  acc[0][1] = __builtin_amdgcn_wmma_f32_16x16x32_bf16(false, a0, false, b1, (short)0, acc[0][1], false, false);
  acc[1][0] = __builtin_amdgcn_wmma_f32_16x16x32_bf16(false, a1, false, b0, (short)0, acc[1][0], false, false);
  acc[1][1] = __builtin_amdgcn_wmma_f32_16x16x32_bf16(false, a1, false, b1, (short)0, acc[1][1], false, false);
}

// ---- LDS staging: 64 rows x 32 K, from v8bf regs, 2 chunks per thread ------
// A: K-permuted store (group g -> g2) so frag reads are contiguous.
__device__ __forceinline__ void stage_a(bf16_t* As_, const v8bf (&ra)[2], int t) {
#pragma unroll
  for (int i = 0; i < 2; ++i) {
    int c = t + i * 128;
    int row = c >> 2, g = c & 3;
    int g2 = ((g & 1) << 1) | (g >> 1);          // 0->0, 1->2, 2->1, 3->3
    *(v8bf*)(As_ + row * APAD + g2 * 8) = ra[i]; // ds_store_b128
  }
}
__device__ __forceinline__ void stage_b(bf16_t* Bt_, const v8bf (&rb)[2], int t) {
#pragma unroll
  for (int i = 0; i < 2; ++i) {
    int c = t + i * 128;
    *(v8bf*)(Bt_ + (c >> 2) * BTP + (((c & 1) << 1) | ((c >> 1) & 1)) * 8) = rb[i]; // ds_store_b128
  }
}

// ---- pipelined GEMM core: double-buffered LDS, one barrier per K-step -----
template <class LA, class LB>
__device__ __forceinline__ void gemm_pipe(int nk, bf16_t* As0, bf16_t* As1,
                                          bf16_t* Bt0, bf16_t* Bt1,
                                          int t, int lane, int wm, int wn,
                                          v8f (&acc)[2][2], LA loadA, LB loadB) {
  v8bf ra[2], rb[2];
  loadA(0, ra); loadB(0, rb);
  stage_a(As0, ra, t); stage_b(Bt0, rb, t);
  __syncthreads();
  for (int k = 0; k < nk; ++k) {
    bf16_t* Ac = (k & 1) ? As1 : As0;
    bf16_t* Bc = (k & 1) ? Bt1 : Bt0;
    if (k + 1 < nk) { loadA(k + 1, ra); loadB(k + 1, rb); }  // overlap VMEM w/ WMMA
    mma_tile(Ac, Bc, wm, wn, lane, acc);
    if (k + 1 < nk) {
      bf16_t* An = (k & 1) ? As0 : As1;
      bf16_t* Bn = (k & 1) ? Bt0 : Bt1;
      stage_a(An, ra, t); stage_b(Bn, rb, t);
      __syncthreads();
    }
  }
}

// D-matrix layout (ISA 7.12.2): VGPR r -> row (lane<16 ? r : 8+r), col lane&15
#define STORE_LOOP(EXPR)                                                     \
  {                                                                          \
    int ms = (lane >> 4) << 3;                                               \
    int nc = lane & 15;                                                      \
    _Pragma("unroll") for (int i = 0; i < 2; ++i)                            \
    _Pragma("unroll") for (int j = 0; j < 2; ++j)                            \
    _Pragma("unroll") for (int r = 0; r < 8; ++r) {                          \
      int rloc = wm * 32 + i * 16 + ms + r;                                  \
      int cloc = wn * 32 + j * 16 + nc;                                      \
      float val = acc[i][j][r];                                              \
      EXPR;                                                                  \
    }                                                                        \
  }

#define GEMM_PROLOGUE()                                                      \
  __shared__ alignas(16) bf16_t As0[64 * APAD];                              \
  __shared__ alignas(16) bf16_t As1[64 * APAD];                              \
  __shared__ alignas(16) bf16_t Bt0[64 * BTP];                               \
  __shared__ alignas(16) bf16_t Bt1[64 * BTP];                               \
  int t = threadIdx.x, lane = t & 31, wave = t >> 5;                         \
  int wm = wave >> 1, wn = wave & 1;                                         \
  v8f acc[2][2] = {};

// ---------------------------------------------------------------------------
// P0a: x (f32) -> x_bf
__global__ __launch_bounds__(256) void k_cvt_x(const float* __restrict__ x,
                                               bf16_t* __restrict__ xbf) {
  int i = (blockIdx.x * 256 + threadIdx.x) * 4;
  float4 v = *(const float4*)&x[i];
  v4bf p; p[0] = (bf16_t)v.x; p[1] = (bf16_t)v.y; p[2] = (bf16_t)v.z; p[3] = (bf16_t)v.w;
  *(v4bf*)&xbf[i] = p;
}

// P0b: w [K][N] f32 -> wT [N][K] bf16
__global__ __launch_bounds__(256) void k_tr_w(const float* __restrict__ w,
                                              bf16_t* __restrict__ wt, int K, int N) {
  int o = blockIdx.x * 256 + threadIdx.x;
  int n = o / K, k = o % K;
  wt[o] = (bf16_t)w[(size_t)k * N + n];
}

// P0c: V part of qkv_bf -> Vt [bh][d][224] bf16, token-contiguous, pad=0
__global__ __launch_bounds__(256) void k_tr_v(const bf16_t* __restrict__ qkvbf,
                                              bf16_t* __restrict__ vt) {
  int o  = blockIdx.x * 256 + threadIdx.x;   // 640*64*224 total
  int kt = o % ALD;
  int d  = (o / ALD) % DD;
  int bh = o / (ALD * DD);
  int b = bh / HH, h = bh % HH;
  vt[o] = (kt < NN) ? qkvbf[(size_t)(b * NN + kt) * QLD + 2 * CC + h * DD + d]
                    : (bf16_t)0.0f;
}

// ---------------------------------------------------------------------------
// K1: qkv_bf = bf16( x @ w_qkv )  [12608,640]x[640,1920]; exact 64-multiples
__global__ __launch_bounds__(128) void k_gemm_qkv(const bf16_t* __restrict__ xbf,
                                                  const bf16_t* __restrict__ wt,
                                                  bf16_t* __restrict__ out) {
  GEMM_PROLOGUE()
  int mb = blockIdx.y * 64, nb = blockIdx.x * 64;
  auto loadA = [&](int kb, v8bf(&ra)[2]) {
#pragma unroll
    for (int i = 0; i < 2; ++i) {
      int c = t + i * 128;
      ra[i] = *(const v8bf*)&xbf[(size_t)(mb + (c >> 2)) * CC + kb * 32 + (c & 3) * 8];
    }
    __builtin_prefetch(&xbf[(size_t)(mb + (t >> 2)) * CC + kb * 32 + 32], 0, 1);
  };
  auto loadB = [&](int kb, v8bf(&rb)[2]) {
#pragma unroll
    for (int i = 0; i < 2; ++i) {
      int c = t + i * 128;
      rb[i] = *(const v8bf*)&wt[(size_t)(nb + (c >> 2)) * CC + kb * 32 + (c & 3) * 8];
    }
    __builtin_prefetch(&wt[(size_t)(nb + (t >> 2)) * CC + kb * 32 + 32], 0, 1);
  };
  gemm_pipe(20, As0, As1, Bt0, Bt1, t, lane, wm, wn, acc, loadA, loadB);
  STORE_LOOP(out[(size_t)(mb + rloc) * QLD + nb + cloc] = (bf16_t)val)
}

// ---------------------------------------------------------------------------
// K2: rel tables: rpk bf16 [q][kt][64]; rpv transposed bf16 [q][d][224], pad=0
__global__ void k_build_rel(const float* __restrict__ tkv, const float* __restrict__ tkh,
                            const float* __restrict__ tvv, const float* __restrict__ tvh,
                            bf16_t* __restrict__ rpk, bf16_t* __restrict__ rpvt) {
  int k = blockIdx.x, q = blockIdx.y, d = threadIdx.x;   // k in 0..223
  if (k >= NN) {
    rpvt[((size_t)q * DD + d) * ALD + k] = (bf16_t)0.0f;
    return;
  }
  int iv = 0, ih = 0;
  if (q > 0 && k > 0) {
    int qi = q - 1, ki = k - 1;
    int dv = ki / 14 - qi / 14;
    int dh = ki % 14 - qi % 14;
    dv = dv < -14 ? -14 : (dv > 14 ? 14 : dv);
    dh = dh < -14 ? -14 : (dh > 14 ? 14 : dh);
    iv = dv + 15; ih = dh + 15;
  }
  rpk[((size_t)q * NN + k) * DD + d]   = (bf16_t)(tkv[iv * DD + d] + tkh[ih * DD + d]);
  rpvt[((size_t)q * DD + d) * ALD + k] = (bf16_t)(tvv[iv * DD + d] + tvh[ih * DD + d]);
}

// ---------------------------------------------------------------------------
// K3: attn[bh,q,k] = SCALE * q.k  (f32 scores; writes k-pad = 0)
__global__ __launch_bounds__(128) void k_attn_qk(const bf16_t* __restrict__ qkv,
                                                 float* __restrict__ attn) {
  GEMM_PROLOGUE()
  int nb = blockIdx.x * 64, mb = blockIdx.y * 64, bh = blockIdx.z;
  int b = bh / HH, h = bh % HH;
  const v8bf z = {};
  auto loadA = [&](int kb, v8bf(&ra)[2]) {
#pragma unroll
    for (int i = 0; i < 2; ++i) {
      int c = t + i * 128;
      int q = mb + (c >> 2);
      ra[i] = (q < NN)
          ? *(const v8bf*)&qkv[(size_t)(b * NN + q) * QLD + h * DD + kb * 32 + (c & 3) * 8]
          : z;
    }
  };
  auto loadB = [&](int kb, v8bf(&rb)[2]) {   // N = k-token row of qkv, K = d contiguous
#pragma unroll
    for (int i = 0; i < 2; ++i) {
      int c = t + i * 128;
      int kt = nb + (c >> 2);
      rb[i] = (kt < NN)
          ? *(const v8bf*)&qkv[(size_t)(b * NN + kt) * QLD + CC + h * DD + kb * 32 + (c & 3) * 8]
          : z;
    }
  };
  gemm_pipe(2, As0, As1, Bt0, Bt1, t, lane, wm, wn, acc, loadA, loadB);
  STORE_LOOP(
    int q = mb + rloc; int k = nb + cloc;
    if (q < NN && k < ALD)
      attn[((size_t)bh * NN + q) * ALD + k] = (k < NN) ? SCALE * val : 0.0f)
}

// ---------------------------------------------------------------------------
// K4: attn[bh,q,k] += SCALE * q.rpk[q,k,:]  (per fixed q: [640,64]x[64,197])
__global__ __launch_bounds__(128) void k_attn_relk(const bf16_t* __restrict__ qkv,
                                                   const bf16_t* __restrict__ rpk,
                                                   float* __restrict__ attn) {
  GEMM_PROLOGUE()
  int nb = blockIdx.x * 64, mb = blockIdx.y * 64, q = blockIdx.z;
  const v8bf z = {};
  auto loadA = [&](int kb, v8bf(&ra)[2]) {
#pragma unroll
    for (int i = 0; i < 2; ++i) {
      int c = t + i * 128;
      int m = mb + (c >> 2);
      int b = m / HH, h = m % HH;
      ra[i] = *(const v8bf*)&qkv[(size_t)(b * NN + q) * QLD + h * DD + kb * 32 + (c & 3) * 8];
    }
  };
  auto loadB = [&](int kb, v8bf(&rb)[2]) {   // N = k-token, K = d contiguous
#pragma unroll
    for (int i = 0; i < 2; ++i) {
      int c = t + i * 128;
      int kt = nb + (c >> 2);
      rb[i] = (kt < NN)
          ? *(const v8bf*)&rpk[((size_t)q * NN + kt) * DD + kb * 32 + (c & 3) * 8]
          : z;
    }
  };
  gemm_pipe(2, As0, As1, Bt0, Bt1, t, lane, wm, wn, acc, loadA, loadB);
  STORE_LOOP(
    int m = mb + rloc; int k = nb + cloc;
    if (k < NN) {
      size_t o = ((size_t)m * NN + q) * ALD + k;
      attn[o] += SCALE * val;
    })
}

// ---------------------------------------------------------------------------
// K5: softmax rows (f32 in, bf16 out, pad zeroed); one wave32 per row
__global__ __launch_bounds__(256) void k_softmax(const float* __restrict__ attn,
                                                 bf16_t* __restrict__ pbf) {
  int row  = blockIdx.x * 8 + (threadIdx.x >> 5);
  int lane = threadIdx.x & 31;
  const float* p = attn + (size_t)row * ALD;
  bf16_t* pb = pbf + (size_t)row * ALD;
  float m = -1e30f;
  for (int i = lane; i < NN; i += 32) m = fmaxf(m, p[i]);
#pragma unroll
  for (int o = 16; o > 0; o >>= 1) m = fmaxf(m, __shfl_xor(m, o, 32));
  float s = 0.0f;
  for (int i = lane; i < NN; i += 32) s += __expf(p[i] - m);
#pragma unroll
  for (int o = 16; o > 0; o >>= 1) s += __shfl_xor(s, o, 32);
  float inv = 1.0f / s;
  for (int i = lane; i < NN; i += 32) pb[i] = (bf16_t)(__expf(p[i] - m) * inv);
  for (int i = NN + lane; i < ALD; i += 32) pb[i] = (bf16_t)0.0f;   // zero k-pad
}

// ---------------------------------------------------------------------------
// K6: ctx[b,q,h,:] = P @ V  (per bh: [197,224]x[224,64], pads are 0)
__global__ __launch_bounds__(128) void k_attn_pv(const bf16_t* __restrict__ pbf,
                                                 const bf16_t* __restrict__ vt,
                                                 float* __restrict__ ctx) {
  GEMM_PROLOGUE()
  int nb = blockIdx.x * 64, mb = blockIdx.y * 64, bh = blockIdx.z;
  int b = bh / HH, h = bh % HH;
  const v8bf z = {};
  auto loadA = [&](int kb, v8bf(&ra)[2]) {
#pragma unroll
    for (int i = 0; i < 2; ++i) {
      int c = t + i * 128;
      int q = mb + (c >> 2);
      ra[i] = (q < NN)
          ? *(const v8bf*)&pbf[((size_t)bh * NN + q) * ALD + kb * 32 + (c & 3) * 8]
          : z;
    }
  };
  auto loadB = [&](int kb, v8bf(&rb)[2]) {   // N = d (row of Vt), K = token contiguous
#pragma unroll
    for (int i = 0; i < 2; ++i) {
      int c = t + i * 128;
      rb[i] = *(const v8bf*)&vt[((size_t)bh * DD + nb + (c >> 2)) * ALD + kb * 32 + (c & 3) * 8];
    }
  };
  gemm_pipe(7, As0, As1, Bt0, Bt1, t, lane, wm, wn, acc, loadA, loadB);
  STORE_LOOP(
    int q = mb + rloc; int d = nb + cloc;
    if (q < NN) ctx[(size_t)(b * NN + q) * CC + h * DD + d] = val)
}

// ---------------------------------------------------------------------------
// K7: ctx_bf = bf16( ctx + P @ rpv[q] )  (per fixed q: [640,224]x[224,64])
__global__ __launch_bounds__(128) void k_attn_relv(const bf16_t* __restrict__ pbf,
                                                   const bf16_t* __restrict__ rpvt,
                                                   const float* __restrict__ ctx,
                                                   bf16_t* __restrict__ ctxbf) {
  GEMM_PROLOGUE()
  int nb = blockIdx.x * 64, mb = blockIdx.y * 64, q = blockIdx.z;
  auto loadA = [&](int kb, v8bf(&ra)[2]) {
#pragma unroll
    for (int i = 0; i < 2; ++i) {
      int c = t + i * 128;
      int m = mb + (c >> 2);
      ra[i] = *(const v8bf*)&pbf[((size_t)m * NN + q) * ALD + kb * 32 + (c & 3) * 8];
    }
  };
  auto loadB = [&](int kb, v8bf(&rb)[2]) {   // N = d (row of rpvt), K = token contiguous
#pragma unroll
    for (int i = 0; i < 2; ++i) {
      int c = t + i * 128;
      rb[i] = *(const v8bf*)&rpvt[((size_t)q * DD + nb + (c >> 2)) * ALD + kb * 32 + (c & 3) * 8];
    }
  };
  gemm_pipe(7, As0, As1, Bt0, Bt1, t, lane, wm, wn, acc, loadA, loadB);
  STORE_LOOP(
    int m = mb + rloc; int d = nb + cloc;
    int b = m / HH; int h = m % HH;
    size_t o = (size_t)(b * NN + q) * CC + h * DD + d;
    ctxbf[o] = (bf16_t)(ctx[o] + val))
}

// ---------------------------------------------------------------------------
// K8: out = ctx_bf @ w_proj + b_proj  [12608,640]x[640,640] (f32 out)
__global__ __launch_bounds__(128) void k_gemm_proj(const bf16_t* __restrict__ ctxbf,
                                                   const bf16_t* __restrict__ wt,
                                                   const float* __restrict__ bias,
                                                   float* __restrict__ out) {
  GEMM_PROLOGUE()
  int mb = blockIdx.y * 64, nb = blockIdx.x * 64;
  auto loadA = [&](int kb, v8bf(&ra)[2]) {
#pragma unroll
    for (int i = 0; i < 2; ++i) {
      int c = t + i * 128;
      ra[i] = *(const v8bf*)&ctxbf[(size_t)(mb + (c >> 2)) * CC + kb * 32 + (c & 3) * 8];
    }
    __builtin_prefetch(&ctxbf[(size_t)(mb + (t >> 2)) * CC + kb * 32 + 32], 0, 1);
  };
  auto loadB = [&](int kb, v8bf(&rb)[2]) {
#pragma unroll
    for (int i = 0; i < 2; ++i) {
      int c = t + i * 128;
      rb[i] = *(const v8bf*)&wt[(size_t)(nb + (c >> 2)) * CC + kb * 32 + (c & 3) * 8];
    }
    __builtin_prefetch(&wt[(size_t)(nb + (t >> 2)) * CC + kb * 32 + 32], 0, 1);
  };
  gemm_pipe(20, As0, As1, Bt0, Bt1, t, lane, wm, wn, acc, loadA, loadB);
  STORE_LOOP(out[(size_t)(mb + rloc) * CC + nb + cloc] = val + bias[nb + cloc])
}

// ---------------------------------------------------------------------------
extern "C" void kernel_launch(void* const* d_in, const int* in_sizes, int n_in,
                              void* d_out, int out_size, void* d_ws, size_t ws_size,
                              hipStream_t stream) {
  (void)in_sizes; (void)n_in; (void)out_size; (void)ws_size;
  const float* x      = (const float*)d_in[0];
  const float* w_qkv  = (const float*)d_in[1];
  const float* w_proj = (const float*)d_in[2];
  const float* b_proj = (const float*)d_in[3];
  const float* tkv    = (const float*)d_in[4];
  const float* tkh    = (const float*)d_in[5];
  const float* tvv    = (const float*)d_in[6];
  const float* tvh    = (const float*)d_in[7];

  char* w8 = (char*)d_ws;
  bf16_t* qkvbf = (bf16_t*)(w8 + OFF_QKVBF);
  float*  attn  = (float*) (w8 + OFF_ATTN);
  bf16_t* pbf   = (bf16_t*)(w8 + OFF_PBF);
  bf16_t* rpk   = (bf16_t*)(w8 + OFF_RPK);
  bf16_t* rpvt  = (bf16_t*)(w8 + OFF_RPVT);
  float*  ctx   = (float*) (w8 + OFF_CTX);
  bf16_t* ctxbf = (bf16_t*)(w8 + OFF_CTXBF);
  bf16_t* xbf   = (bf16_t*)(w8 + OFF_XBF);
  bf16_t* wqt   = (bf16_t*)(w8 + OFF_WQT);
  bf16_t* wpt   = (bf16_t*)(w8 + OFF_WPT);
  bf16_t* vt    = (bf16_t*)(w8 + OFF_VT);
  float*  out   = (float*)d_out;

  // P0: bf16 conversions / transposes (one pass each, all L2-resident)
  k_cvt_x<<<MT * CC / (256 * 4), 256, 0, stream>>>(x, xbf);            // 7880 blocks
  k_tr_w <<<1920 * CC / 256, 256, 0, stream>>>(w_qkv, wqt, CC, 1920);  // 4800 blocks
  k_tr_w <<<CC * CC / 256, 256, 0, stream>>>(w_proj, wpt, CC, CC);     // 1600 blocks
  // K2: rel tables (independent of qkv)
  k_build_rel<<<dim3(ALD, NN), DD, 0, stream>>>(tkv, tkh, tvv, tvh, rpk, rpvt);
  // K1: qkv GEMM -> bf16 (M=12608, N=1920, K=640)
  k_gemm_qkv<<<dim3(30, 197), 128, 0, stream>>>(xbf, wqt, qkvbf);
  // P0c: V^T extraction (bh,d,token) zero-padded
  k_tr_v<<<640 * DD * ALD / 256, 256, 0, stream>>>(qkvbf, vt);         // 35840 blocks
  // K3: q@k^T per (b,h); zeroes k-pad
  k_attn_qk<<<dim3(4, 4, 640), 128, 0, stream>>>(qkvbf, attn);
  // K4: += q@rpk^T per fixed q, batched over (b,h)
  k_attn_relk<<<dim3(4, 10, NN), 128, 0, stream>>>(qkvbf, rpk, attn);
  // K5: softmax -> bf16 P (padded)
  k_softmax<<<15760, 256, 0, stream>>>(attn, pbf);
  // K6: P@V per (b,h)
  k_attn_pv<<<dim3(1, 4, 640), 128, 0, stream>>>(pbf, vt, ctx);
  // K7: ctx_bf = bf16(ctx + P@rpv) per fixed q
  k_attn_relv<<<dim3(1, 10, NN), 128, 0, stream>>>(pbf, rpvt, ctx, ctxbf);
  // K8: projection GEMM + bias
  k_gemm_proj<<<dim3(10, 197), 128, 0, stream>>>(ctxbf, wpt, b_proj, out);
}
